// NeuralMapCell_20615843021177
// MI455X (gfx1250) — matrix operand order repaired and checked by reading, not compiled, not run on hardware
//
#include <hip/hip_runtime.h>

// ---------------------------------------------------------------------------
// Types for CDNA5 WMMA
// ---------------------------------------------------------------------------
typedef __attribute__((ext_vector_type(16))) __bf16 v16bf;
typedef __attribute__((ext_vector_type(8)))  float  v8f;

union ABfrag {
    v16bf bf;
    uint4 q[2];
};

static __device__ __forceinline__ unsigned short f2bf(float f) {
    unsigned int u = __float_as_uint(f);
    unsigned int r = (u + 0x7FFFu + ((u >> 16) & 1u)) >> 16;
    return (unsigned short)r;
}

static __device__ __forceinline__ void mask_frag(ABfrag& a, unsigned m) {
    a.q[0].x &= m; a.q[0].y &= m; a.q[0].z &= m; a.q[0].w &= m;
    a.q[1].x &= m; a.q[1].y &= m; a.q[1].z &= m; a.q[1].w &= m;
}

// ---------------------------------------------------------------------------
// Problem constants
// ---------------------------------------------------------------------------
#define UNITS 256
#define HW    4096           // 64*64
#define FLAT  238144         // 64*61*61

// workspace offsets (bytes)
#define OFF_A0     0u                        // 4096*256 bf16  = 2,097,152
#define OFF_W1T    2097152u                  // 9*32*256 bf16  =   147,456
#define OFF_W2T    2244608u                  // 9*64*32  bf16  =    36,864
#define OFF_C1     2281472u                  // 4096*32  bf16  =   262,144
#define OFF_C2     2543616u                  // 64*4096  f32   = 1,048,576
#define OFF_RPRE   3592192u                  // 128 f32
#define OFF_MEMNEW 3592704u                  // 256 f32

// ---------------------------------------------------------------------------
// Prep: f32 memory -> bf16 (pixel-major, 4096 x 256)
// ---------------------------------------------------------------------------
__global__ void k_prep_mem(const float* __restrict__ mem, unsigned short* __restrict__ A0) {
    int i = blockIdx.x * blockDim.x + threadIdx.x;   // 0 .. 1048575
    A0[i] = f2bf(mem[i]);
}

// Prep: transpose conv weights to N-major bf16; zero r_pre accumulator.
// conv_kernel1: (3,3,256,32) HWIO -> w1t[tap][oc][c]
// conv_kernel2: (3,3,32,64)  HWIO -> w2t[tap][oc][ic]
__global__ void k_prep_w(const float* __restrict__ K1, const float* __restrict__ K2,
                         unsigned short* __restrict__ W1t, unsigned short* __restrict__ W2t,
                         float* __restrict__ r_pre) {
    int i = blockIdx.x * blockDim.x + threadIdx.x;
    if (i < 9 * 32 * 256) {
        int tap = i / 8192, r = i % 8192;
        int oc = r / 256, c = r % 256;
        W1t[i] = f2bf(K1[(tap * 256 + c) * 32 + oc]);
    }
    if (i < 9 * 64 * 32) {
        int tap = i / 2048, r = i % 2048;
        int oc = r / 32, ic = r % 32;
        W2t[i] = f2bf(K2[(tap * 32 + ic) * 64 + oc]);
    }
    if (i < 128) r_pre[i] = 0.0f;
}

// ---------------------------------------------------------------------------
// conv1: SAME 3x3, 256 -> 32 channels, implicit GEMM as 9 shifted GEMMs.
// One wave per 16-pixel M tile; first 8 KB (oc 0..15) of the 16 KB per-tap weight tile staged in LDS and
// shared by the 8 waves of the block. A-fragment edge handling is branchless
// (clamped address + AND mask) so EXEC stays all-ones through the WMMAs.
// ---------------------------------------------------------------------------
__global__ void k_conv1(const unsigned short* __restrict__ A0,
                        const unsigned short* __restrict__ W1t,
                        unsigned short* __restrict__ C1) {
    __shared__ unsigned short w1s[32 * 256];      // 16 KB per-tap weight tile

    int tid  = threadIdx.x;
    int wave = (blockIdx.x * blockDim.x + tid) >> 5;   // 0..255
    int lane = tid & 31;
    int p0 = wave * 16;
    int h  = p0 >> 6;
    int w0 = p0 & 63;
    int mrow = lane & 15;
    int half = lane >> 4;

    v8f acc0 = {};
    v8f acc1 = {};

    for (int tap = 0; tap < 9; ++tap) {
        // stage oc rows 0..15 (8 KB) of this tap's 32x256 bf16 weight tile into LDS (512 x b128)
        __syncthreads();
        {
            const uint4* src = (const uint4*)(W1t + (size_t)tap * 8192);
            uint4* dst = (uint4*)w1s;
            dst[tid]       = src[tid];
            dst[tid + 256] = src[tid + 256];
        }
        __syncthreads();

        int dh = tap / 3, dw = tap % 3;
        int hh = h + dh - 1;
        int ww = w0 + mrow + dw - 1;
        bool av = (hh >= 0) && (hh <= 63) && (ww >= 0) && (ww <= 63);
        unsigned msk = av ? 0xFFFFFFFFu : 0u;
        int hc = hh < 0 ? 0 : (hh > 63 ? 63 : hh);
        int wc = ww < 0 ? 0 : (ww > 63 ? 63 : ww);
        const unsigned short* arow  = A0 + (size_t)(hc * 64 + wc) * 256;
        const unsigned short* brow0 = w1s + (lane & 15) * 256;
        const unsigned short* brow1 = brow0 + 16 * 256;

        for (int kc = 0; kc < 8; ++kc) {
            ABfrag a;
            int ka = kc * 32 + half * 8;
            a.q[0] = *(const uint4*)(arow + ka);
            a.q[1] = *(const uint4*)(arow + ka + 16);
            mask_frag(a, msk);

            int kb = kc * 32 + half * 8;
            ABfrag b0, b1;
            b0.q[0] = *(const uint4*)(brow0 + kb);
            b0.q[1] = *(const uint4*)(brow0 + kb + 16);
            b1.q[0] = *(const uint4*)(brow1 + kb);
            b1.q[1] = *(const uint4*)(brow1 + kb + 16);

            acc0 = __builtin_amdgcn_wmma_f32_16x16x32_bf16(false, a.bf, false, b0.bf,
                                                           (short)0, acc0, false, false);
            acc1 = __builtin_amdgcn_wmma_f32_16x16x32_bf16(false, a.bf, false, b1.bf,
                                                           (short)0, acc1, false, false);
        }
    }
    // store bf16, pixel-major (4096 x 32)
    for (int r = 0; r < 8; ++r) {
        int p = p0 + half * 8 + r;
        C1[p * 32 + (lane & 15)]      = f2bf(acc0[r]);
        C1[p * 32 + 16 + (lane & 15)] = f2bf(acc1[r]);
    }
}

// ---------------------------------------------------------------------------
// conv2: VALID 3x3, 32 -> 64 channels; output stored channel-major f32 on the
// full 64x64 grid (only h,w < 62 is valid and consumed downstream).
// First 2 KB (oc 0..31) of the 4 KB per-tap weight tile staged in LDS; branchless A-edge masking.
// ---------------------------------------------------------------------------
__global__ void k_conv2(const unsigned short* __restrict__ C1,
                        const unsigned short* __restrict__ W2t,
                        float* __restrict__ C2cm) {
    __shared__ unsigned short w2s[64 * 32];       // 4 KB per-tap weight tile

    int tid  = threadIdx.x;
    int wave = (blockIdx.x * blockDim.x + tid) >> 5;   // 0..255
    int lane = tid & 31;
    int p0 = wave * 16;
    int h  = p0 >> 6;
    int w0 = p0 & 63;
    int mrow = lane & 15;
    int half = lane >> 4;

    v8f acc[4] = {v8f{}, v8f{}, v8f{}, v8f{}};

    for (int tap = 0; tap < 9; ++tap) {
        __syncthreads();
        if (tid < 128) {
            ((uint4*)w2s)[tid] = ((const uint4*)(W2t + (size_t)tap * 2048))[tid];
        }
        __syncthreads();

        int dh = tap / 3, dw = tap % 3;
        int hh = h + dh;
        int ww = w0 + mrow + dw;
        bool av = (hh <= 63) && (ww <= 63);
        unsigned msk = av ? 0xFFFFFFFFu : 0u;
        int hc = hh > 63 ? 63 : hh;
        int wc = ww > 63 ? 63 : ww;
        const unsigned short* arow = C1 + (size_t)(hc * 64 + wc) * 32;

        ABfrag a;
        a.q[0] = *(const uint4*)(arow + half * 8);
        a.q[1] = *(const uint4*)(arow + half * 8 + 16);
        mask_frag(a, msk);

        int kb = half * 8;
        for (int nt = 0; nt < 4; ++nt) {
            const unsigned short* brow = w2s + (nt * 16 + (lane & 15)) * 32 + kb;
            ABfrag b;
            b.q[0] = *(const uint4*)(brow);
            b.q[1] = *(const uint4*)(brow + 16);
            acc[nt] = __builtin_amdgcn_wmma_f32_16x16x32_bf16(false, a.bf, false, b.bf,
                                                              (short)0, acc[nt], false, false);
        }
    }
    for (int nt = 0; nt < 4; ++nt)
        for (int r = 0; r < 8; ++r) {
            int oc = nt * 16 + (lane & 15);
            int p  = p0 + half * 8 + r;
            C2cm[(size_t)oc * HW + p] = acc[nt][r];
        }
}

// ---------------------------------------------------------------------------
// fused 2x2 avg pool (stride 1) + flat @ conv_dense1 -> r_pre[128]
// One wave per flat index f; lane j covers output cols 4j..4j+3.
// Byte-dominant stage: streams conv_dense1 (122 MB) once, coalesced float4,
// with software prefetch of the next row (global_prefetch_b8).
// ---------------------------------------------------------------------------
__global__ void k_pool_dense1(const float* __restrict__ C2cm,
                              const float* __restrict__ D1,
                              float* __restrict__ r_pre) {
    int gwave  = (blockIdx.x * blockDim.x + threadIdx.x) >> 5;
    int lane   = threadIdx.x & 31;
    int nwaves = (gridDim.x * blockDim.x) >> 5;

    float a0 = 0.f, a1 = 0.f, a2 = 0.f, a3 = 0.f;
    for (int f = gwave; f < FLAT; f += nwaves) {
        if (f + nwaves < FLAT)
            __builtin_prefetch(D1 + (size_t)(f + nwaves) * 128 + lane * 4, 0, 1);
        unsigned oc  = (unsigned)f / 3721u;
        unsigned rem = (unsigned)f % 3721u;
        unsigned oh  = rem / 61u, ow = rem % 61u;
        const float* c2 = C2cm + (size_t)oc * HW + oh * 64 + ow;
        float pv = 0.25f * (c2[0] + c2[1] + c2[64] + c2[65]);
        float4 dv = *(const float4*)(D1 + (size_t)f * 128 + lane * 4);
        a0 += pv * dv.x; a1 += pv * dv.y; a2 += pv * dv.z; a3 += pv * dv.w;
    }
    atomicAdd(r_pre + lane * 4 + 0, a0);
    atomicAdd(r_pre + lane * 4 + 1, a1);
    atomicAdd(r_pre + lane * 4 + 2, a2);
    atomicAdd(r_pre + lane * 4 + 3, a3);
}

// ---------------------------------------------------------------------------
// head: r_t, q_t, attention scores + softmax, c_t, s_t, importances, cell
// update. Single 256-thread block; everything is M=1 GEMV.
// ---------------------------------------------------------------------------
__global__ void k_head(const float* __restrict__ inputs, const float* __restrict__ memory,
                       const float* __restrict__ D2, const float* __restrict__ CK,
                       const float* __restrict__ REC, const float* __restrict__ r_pre,
                       const int* __restrict__ px, const int* __restrict__ py,
                       float* __restrict__ d_out, float* __restrict__ mem_new) {
    __shared__ float r_pre_l[128];
    __shared__ float q_in[384];          // [inputs(128) ; r_t(256)]
    __shared__ float q_t[256];
    __shared__ float sc[4096];
    __shared__ float red[256];
    __shared__ float s_t[256];
    __shared__ float c_t_l[256];
    __shared__ float dvec[256];          // mem_t - s_t

    int t = threadIdx.x;
    if (t < 128) { r_pre_l[t] = r_pre[t]; q_in[t] = inputs[t]; }
    __syncthreads();

    // r_t = r_pre @ conv_dense2
    {
        float a = 0.f;
        for (int k = 0; k < 128; ++k) a += r_pre_l[k] * D2[k * 256 + t];
        q_in[128 + t] = a;
        d_out[256 + t] = a;              // r_t output
    }
    __syncthreads();

    // q_t = [inputs, r_t] @ context_kernel
    {
        float a = 0.f;
        for (int k = 0; k < 384; ++k) a += q_in[k] * CK[k * 256 + t];
        q_t[t] = a;
    }
    __syncthreads();

    // scores[p] = q_t . memory[p,:]   (m2[c][p] == memory[p][c])
    float lmax = -3.4e38f;
    for (int p = t; p < HW; p += 256) {
        const float* mr = memory + (size_t)p * 256;
        float s = 0.f;
        for (int c = 0; c < 256; c += 4) {
            float4 m4 = *(const float4*)(mr + c);
            s += q_t[c] * m4.x + q_t[c + 1] * m4.y + q_t[c + 2] * m4.z + q_t[c + 3] * m4.w;
        }
        sc[p] = s;
        lmax = fmaxf(lmax, s);
    }
    red[t] = lmax; __syncthreads();
    for (int s = 128; s > 0; s >>= 1) {
        if (t < s) red[t] = fmaxf(red[t], red[t + s]);
        __syncthreads();
    }
    float gmax = red[0]; __syncthreads();

    float lsum = 0.f;
    for (int p = t; p < HW; p += 256) {
        float e = __expf(sc[p] - gmax);
        sc[p] = e;
        lsum += e;
    }
    red[t] = lsum; __syncthreads();
    for (int s = 128; s > 0; s >>= 1) {
        if (t < s) red[t] = red[t] + red[t + s];
        __syncthreads();
    }
    float inv = 1.0f / red[0]; __syncthreads();

    // c_t = at @ m2.T   -> c_t[c] = sum_p softmax[p] * memory[p][c]
    {
        float a = 0.f;
        for (int p = 0; p < HW; ++p) a += sc[p] * memory[(size_t)p * 256 + t];
        a *= inv;
        c_t_l[t] = a;
        d_out[t] = a;                    // c_t output
    }
    // s_t = inputs @ rec[:128]
    {
        float a = 0.f;
        for (int k = 0; k < 128; ++k) a += q_in[k] * REC[k * 256 + t];
        s_t[t] = a;
    }
    __syncthreads();

    // importances
    red[t] = s_t[t] * q_in[128 + t]; __syncthreads();          // global: s_t . r_t
    for (int s = 128; s > 0; s >>= 1) { if (t < s) red[t] += red[t + s]; __syncthreads(); }
    float gimp = red[0]; __syncthreads();
    red[t] = s_t[t] * c_t_l[t]; __syncthreads();               // local:  s_t . c_t
    for (int s = 128; s > 0; s >>= 1) { if (t < s) red[t] += red[t + s]; __syncthreads(); }
    float limp = red[0]; __syncthreads();

    // cell update: mem_t += frac * ((mem_t - s_t) @ rec[128:])
    int pxy = (*px) * 64 + (*py);
    float mt = memory[(size_t)pxy * 256 + t];
    dvec[t] = mt - s_t[t];
    float frac = limp / (limp + gimp);
    __syncthreads();
    float delta = 0.f;
    for (int k = 0; k < 256; ++k) delta += dvec[k] * REC[(128 + k) * 256 + t];
    mem_new[t] = mt + frac * delta;
}

// ---------------------------------------------------------------------------
// new_mem: LDS-tiled transpose of memory (4096 x 256 -> 256 x 4096) with the
// column at pixel (x,y) replaced by the updated cell.
// ---------------------------------------------------------------------------
__global__ void k_writemem(const float* __restrict__ memory, const float* __restrict__ mem_new,
                           const int* __restrict__ px, const int* __restrict__ py,
                           float* __restrict__ out_mem) {
    __shared__ float tile[32][33];
    int p0 = blockIdx.x * 32;   // pixel tile
    int c0 = blockIdx.y * 32;   // channel tile
    int lc = threadIdx.x & 31;
    int lr = threadIdx.x >> 5;  // 0..7
    for (int r = 0; r < 4; ++r) {
        int pr = lr + r * 8;
        tile[pr][lc] = memory[(size_t)(p0 + pr) * 256 + c0 + lc];
    }
    __syncthreads();
    int pxy = (*px) * 64 + (*py);
    for (int r = 0; r < 4; ++r) {
        int cr = lr + r * 8;
        int p  = p0 + lc;
        float v = tile[lc][cr];
        if (p == pxy) v = mem_new[c0 + cr];
        out_mem[(size_t)(c0 + cr) * HW + p] = v;
    }
}

// ---------------------------------------------------------------------------
// launch
// ---------------------------------------------------------------------------
extern "C" void kernel_launch(void* const* d_in, const int* in_sizes, int n_in,
                              void* d_out, int out_size, void* d_ws, size_t ws_size,
                              hipStream_t stream) {
    const float* inputs = (const float*)d_in[0];
    const float* memory = (const float*)d_in[1];
    const float* ck1    = (const float*)d_in[2];
    const float* ck2    = (const float*)d_in[3];
    const float* cd1    = (const float*)d_in[4];
    const float* cd2    = (const float*)d_in[5];
    const float* ctk    = (const float*)d_in[6];
    const float* rec    = (const float*)d_in[7];
    const int*   px     = (const int*)d_in[8];
    const int*   py     = (const int*)d_in[9];
    float* out = (float*)d_out;

    char* ws = (char*)d_ws;
    unsigned short* A0   = (unsigned short*)(ws + OFF_A0);
    unsigned short* W1t  = (unsigned short*)(ws + OFF_W1T);
    unsigned short* W2t  = (unsigned short*)(ws + OFF_W2T);
    unsigned short* C1   = (unsigned short*)(ws + OFF_C1);
    float* C2cm          = (float*)(ws + OFF_C2);
    float* r_pre         = (float*)(ws + OFF_RPRE);
    float* mem_new       = (float*)(ws + OFF_MEMNEW);

    k_prep_mem<<<4096, 256, 0, stream>>>(memory, A0);
    k_prep_w<<<288, 256, 0, stream>>>(ck1, ck2, W1t, W2t, r_pre);
    k_conv1<<<32, 256, 0, stream>>>(A0, W1t, C1);
    k_conv2<<<32, 256, 0, stream>>>(C1, W2t, C2cm);
    k_pool_dense1<<<256, 256, 0, stream>>>(C2cm, cd1, r_pre);
    k_head<<<1, 256, 0, stream>>>(inputs, memory, cd2, ctk, rec, r_pre, px, py, out, mem_new);
    k_writemem<<<dim3(128, 8), 256, 0, stream>>>(memory, mem_new, px, py, out + 512);
}
